// TextLSTM_66597762892363
// MI455X (gfx1250) — matrix operand order repaired and checked by reading, not compiled, not run on hardware
//
#include <hip/hip_runtime.h>
#include <cmath>

typedef __bf16 bf16;
typedef __attribute__((ext_vector_type(16))) __bf16 v16bf;
typedef __attribute__((ext_vector_type(8)))  __bf16 v8bf;
typedef __attribute__((ext_vector_type(8)))  float  v8f;
typedef __attribute__((ext_vector_type(4)))  unsigned int v4u;
typedef __attribute__((ext_vector_type(8)))  int v8i;
typedef __attribute__((ext_vector_type(4)))  int v4i;

#define S_LEN 1024
#define BATCH 64

// ---------------------------------------------------------------------------
// TDM: stage a contiguous row-major panel (rows x width_elems of bf16) from
// global memory into LDS at byte offset lds_off. Descriptor per ISA 8.3-8.5.
// Issue from a single wave; EXEC is ignored by TDM. 6-arg builtin variant.
// ---------------------------------------------------------------------------
__device__ __forceinline__ void tdm_load_panel(unsigned lds_off,
                                               const bf16* gptr,
                                               unsigned width_elems,
                                               unsigned rows) {
  unsigned long long ga = (unsigned long long)(uintptr_t)gptr;
  v4u g0;
  g0[0] = 1u;                                              // count=1, user mode
  g0[1] = lds_off;                                         // LDS byte address
  g0[2] = (unsigned)(ga & 0xffffffffu);                    // global_addr lo
  g0[3] = (unsigned)((ga >> 32) & 0x01ffffffu) | (2u << 30); // addr hi | type=2
  v8i g1;
  g1[0] = (int)(1u << 16);                                 // data_size=1 (2 bytes)
  g1[1] = (int)((width_elems & 0xffffu) << 16);            // tensor_dim0[15:0]
  g1[2] = (int)(((width_elems >> 16) & 0xffffu) |
                ((rows & 0xffffu) << 16));                 // dim0 hi | dim1 lo
  g1[3] = (int)(((rows >> 16) & 0xffffu) |
                ((width_elems & 0xffffu) << 16));          // dim1 hi | tile_dim0
  g1[4] = (int)(rows & 0xffffu);                           // tile_dim1 | tile_dim2=0
  g1[5] = (int)width_elems;                                // dim0_stride lo32
  g1[6] = 0;
  g1[7] = 0;
  v4i z4 = {0, 0, 0, 0};
  v8i z8 = {0, 0, 0, 0, 0, 0, 0, 0};
  __builtin_amdgcn_tensor_load_to_lds(g0, g1, z4, z4, z8, 0);
}

// ---------------------------------------------------------------------------
// WMMA fragment loaders (wave32, 16x16x32 bf16), contiguous k = 16*half + s in A and B
// ---------------------------------------------------------------------------
__device__ __forceinline__ v16bf load_a_tile(const bf16* __restrict__ A, int lda,
                                             int lane, int k0) {
  const int half = (lane >> 4) & 1;
  const int m    = lane & 15;
  const bf16* p = A + (size_t)m * lda + k0 + half * 16;
  v8bf lo = *(const v8bf*)(p);        // 16B
  v8bf hi = *(const v8bf*)(p + 8);    // 16B
  v16bf a;
#pragma unroll
  for (int i = 0; i < 8; ++i) { a[i] = lo[i]; a[i + 8] = hi[i]; }
  return a;
}

__device__ __forceinline__ v16bf load_b_tile(const bf16* __restrict__ W, int ldw,
                                             int lane, int n0, int k0) {
  const int half = (lane >> 4) & 1;
  const int n    = lane & 15;
  return *(const v16bf*)(W + (size_t)(n0 + n) * ldw + k0 + half * 16); // 32B
}

// ---------------------------------------------------------------------------
// gates = [x_t | h_{s-1}] @ [Wih | Whh]^T + bih + bhh      (64 x 4H)
// Block = 8 waves; block stages x_t (64xE) and h_{s-1} (64xH) panels into LDS
// via TDM. Each wave owns a 64x16 output block (all 4 M-tiles, one N-tile):
// per k-step ONE B fragment from global (unique operand, loaded exactly once
// per block) feeds 4 WMMAs whose A fragments come from LDS. B is manually
// double-buffered so the next tile's loads overlap the current 4 WMMAs.
// ---------------------------------------------------------------------------
__global__ void lstm_gates_wmma(const bf16* __restrict__ xt, int E,
                                const bf16* __restrict__ hprev, int H,
                                const bf16* __restrict__ wih,
                                const bf16* __restrict__ whh,
                                const float* __restrict__ bih,
                                const float* __restrict__ bhh,
                                float* __restrict__ gates) {
  extern __shared__ bf16 smem[];              // [64*E] x-panel, then [64*H] h-panel
  bf16* sx = smem;
  bf16* sh = smem + (size_t)BATCH * E;

  const int N4   = 4 * H;
  const int wid  = (blockIdx.x * blockDim.x + threadIdx.x) >> 5;
  const int lane = threadIdx.x & 31;
  const int n0   = wid << 4;                  // one 16-col N-tile per wave

  // wave 0 DMAs the activation panels into LDS
  if (threadIdx.x < 32) {
    tdm_load_panel(0u, xt, (unsigned)E, BATCH);
    if (hprev != nullptr)
      tdm_load_panel((unsigned)(BATCH * E * sizeof(bf16)), hprev, (unsigned)H, BATCH);
    __builtin_amdgcn_s_wait_tensorcnt(0);
  }
  __syncthreads();

  if (n0 >= N4) return;                       // wave-uniform

  v8f acc0 = {}, acc1 = {}, acc2 = {}, acc3 = {};

  // ---- input projection: K = E, A panels from LDS, B double-buffered ----
  {
    v16bf b = load_b_tile(wih, E, lane, n0, 0);
    for (int k0 = 0; k0 < E; k0 += 32) {
      v16bf a0 = load_a_tile(sx,                    E, lane, k0);
      v16bf a1 = load_a_tile(sx + (size_t)16 * E,   E, lane, k0);
      v16bf a2 = load_a_tile(sx + (size_t)32 * E,   E, lane, k0);
      v16bf a3 = load_a_tile(sx + (size_t)48 * E,   E, lane, k0);
      v16bf bn = b;
      if (k0 + 32 < E) bn = load_b_tile(wih, E, lane, n0, k0 + 32);
      acc0 = __builtin_amdgcn_wmma_f32_16x16x32_bf16(false, a0, false, b, (short)0, acc0, false, false);
      acc1 = __builtin_amdgcn_wmma_f32_16x16x32_bf16(false, a1, false, b, (short)0, acc1, false, false);
      acc2 = __builtin_amdgcn_wmma_f32_16x16x32_bf16(false, a2, false, b, (short)0, acc2, false, false);
      acc3 = __builtin_amdgcn_wmma_f32_16x16x32_bf16(false, a3, false, b, (short)0, acc3, false, false);
      b = bn;
    }
  }
  // ---- recurrent projection: K = H (skipped at s==0 where h_prev == 0) ----
  if (hprev != nullptr) {
    v16bf b = load_b_tile(whh, H, lane, n0, 0);
    for (int k0 = 0; k0 < H; k0 += 32) {
      v16bf a0 = load_a_tile(sh,                    H, lane, k0);
      v16bf a1 = load_a_tile(sh + (size_t)16 * H,   H, lane, k0);
      v16bf a2 = load_a_tile(sh + (size_t)32 * H,   H, lane, k0);
      v16bf a3 = load_a_tile(sh + (size_t)48 * H,   H, lane, k0);
      v16bf bn = b;
      if (k0 + 32 < H) bn = load_b_tile(whh, H, lane, n0, k0 + 32);
      acc0 = __builtin_amdgcn_wmma_f32_16x16x32_bf16(false, a0, false, b, (short)0, acc0, false, false);
      acc1 = __builtin_amdgcn_wmma_f32_16x16x32_bf16(false, a1, false, b, (short)0, acc1, false, false);
      acc2 = __builtin_amdgcn_wmma_f32_16x16x32_bf16(false, a2, false, b, (short)0, acc2, false, false);
      acc3 = __builtin_amdgcn_wmma_f32_16x16x32_bf16(false, a3, false, b, (short)0, acc3, false, false);
      b = bn;
    }
  }

  // C/D layout: VGPR v -> row (v + 8*half), col = lane&15
  const int half = lane >> 4;
  const int n    = lane & 15;
  const int col  = n0 + n;
  const float bias = bih[col] + bhh[col];
  v8f accs[4] = {acc0, acc1, acc2, acc3};
#pragma unroll
  for (int i = 0; i < 4; ++i) {
    const int m0 = i << 4;
#pragma unroll
    for (int v = 0; v < 8; ++v) {
      gates[(size_t)(m0 + half * 8 + v) * N4 + col] = accs[i][v] + bias;
    }
  }
}

// ---------------------------------------------------------------------------
// Pointwise LSTM cell update. PyTorch gate order: i, f, g, o.
// ---------------------------------------------------------------------------
__global__ void lstm_pointwise(const float* __restrict__ gates,
                               float* __restrict__ cstate,
                               bf16* __restrict__ hout,
                               float* __restrict__ last,
                               int H, int first) {
  int idx = blockIdx.x * blockDim.x + threadIdx.x;
  if (idx >= BATCH * H) return;
  int b = idx / H, j = idx - b * H;
  const float* gr = gates + (size_t)b * 4 * H;
  float gi = gr[j];
  float gf = gr[H + j];
  float gg = gr[2 * H + j];
  float go = gr[3 * H + j];
  float i = 1.f / (1.f + __expf(-gi));
  float f = 1.f / (1.f + __expf(-gf));
  float g = tanhf(gg);
  float o = 1.f / (1.f + __expf(-go));
  float cp = first ? 0.f : cstate[idx];
  float c = f * cp + i * g;
  float h = o * tanhf(c);
  cstate[idx] = c;
  hout[(size_t)b * H + j] = (bf16)h;
  if (last != nullptr && b == BATCH - 1) last[j] = h;
}

// ---------------------------------------------------------------------------
__global__ void embed_bf16(const int* __restrict__ tokens,
                           const float* __restrict__ emb,
                           bf16* __restrict__ x, int total) {
  int idx = blockIdx.x * blockDim.x + threadIdx.x;
  if (idx >= total) return;
  int r = idx >> 9;          // row = s*64 + b  (E = 512)
  int e = idx & 511;
  int tok = tokens[r];
  x[idx] = (bf16)emb[(size_t)tok * 512 + e];
}

__global__ void f32_to_bf16(const float* __restrict__ src, bf16* __restrict__ dst, int n) {
  int idx = blockIdx.x * blockDim.x + threadIdx.x;
  if (idx < n) dst[idx] = (bf16)src[idx];
}

// out[s][o] = last[s] . Wlin[o] + blin[o]   (1024 x 7, K=256; trivial)
__global__ void final_linear(const float* __restrict__ last,
                             const float* __restrict__ Wlin,
                             const float* __restrict__ blin,
                             float* __restrict__ out) {
  int idx = blockIdx.x * blockDim.x + threadIdx.x;
  if (idx >= S_LEN * 7) return;
  int s = idx / 7, o = idx - s * 7;
  const float* lr = last + (size_t)s * 256;
  const float* wr = Wlin + (size_t)o * 256;
  float acc = blin[o];
  for (int k = 0; k < 256; ++k) acc += lr[k] * wr[k];
  out[idx] = acc;
}

// ---------------------------------------------------------------------------
extern "C" void kernel_launch(void* const* d_in, const int* in_sizes, int n_in,
                              void* d_out, int out_size, void* d_ws, size_t ws_size,
                              hipStream_t stream) {
  (void)in_sizes; (void)n_in; (void)out_size; (void)ws_size;

  const int*   tokens = (const int*)d_in[0];
  const float* emb    = (const float*)d_in[1];
  const float* Wih[3] = {(const float*)d_in[2], (const float*)d_in[6],  (const float*)d_in[10]};
  const float* Whh[3] = {(const float*)d_in[3], (const float*)d_in[7],  (const float*)d_in[11]};
  const float* bihp[3]= {(const float*)d_in[4], (const float*)d_in[8],  (const float*)d_in[12]};
  const float* bhhp[3]= {(const float*)d_in[5], (const float*)d_in[9],  (const float*)d_in[13]};
  const float* Wlin   = (const float*)d_in[14];
  const float* blin   = (const float*)d_in[15];
  float* out = (float*)d_out;

  const int E[3] = {512, 1024, 512};   // layer input widths
  const int H[3] = {1024, 512, 256};   // layer hidden widths
  const size_t SB = (size_t)S_LEN * BATCH;   // 65536 rows

  // allow up to 192KB dynamic LDS for the gates kernel (WGP has 320KB);
  // idempotent, capture-safe (not a stream op)
  (void)hipFuncSetAttribute(reinterpret_cast<const void*>(lstm_gates_wmma),
                            hipFuncAttributeMaxDynamicSharedMemorySize, 196608);

  // ---- workspace carve-out (~215 MB total) ----
  char* w = (char*)d_ws;
  size_t off = 0;
  auto wsalloc = [&](size_t bytes) -> void* {
    void* p = w + off;
    off = (off + bytes + 255) & ~(size_t)255;
    return p;
  };
  bf16* bufA = (bf16*)wsalloc(SB * 512  * sizeof(bf16));  // x / hs1 (width<=512)
  bf16* bufB = (bf16*)wsalloc(SB * 1024 * sizeof(bf16));  // hs0 / hs2 (width<=1024)
  bf16* wihb[3];
  bf16* whhb[3];
  for (int l = 0; l < 3; ++l) {
    wihb[l] = (bf16*)wsalloc((size_t)4 * H[l] * E[l] * sizeof(bf16));
    whhb[l] = (bf16*)wsalloc((size_t)4 * H[l] * H[l] * sizeof(bf16));
  }
  float* gates  = (float*)wsalloc((size_t)BATCH * 4096 * sizeof(float));
  float* cstate = (float*)wsalloc((size_t)BATCH * 1024 * sizeof(float));
  float* last   = (float*)wsalloc((size_t)S_LEN * 256  * sizeof(float));

  // ---- weights -> bf16 (fp32 masters untouched) ----
  for (int l = 0; l < 3; ++l) {
    int nih = 4 * H[l] * E[l];
    int nhh = 4 * H[l] * H[l];
    f32_to_bf16<<<(nih + 255) / 256, 256, 0, stream>>>(Wih[l], wihb[l], nih);
    f32_to_bf16<<<(nhh + 255) / 256, 256, 0, stream>>>(Whh[l], whhb[l], nhh);
  }

  // ---- embedding gather -> bf16 activations ----
  {
    int total = (int)(SB * 512);
    embed_bf16<<<(total + 255) / 256, 256, 0, stream>>>(tokens, emb, bufA, total);
  }

  // ---- three stacked LSTM layers, sequential scan over 1024 steps ----
  bf16* layer_in[3]  = {bufA, bufB, bufA};
  bf16* layer_out[3] = {bufB, bufA, bufB};
  for (int l = 0; l < 3; ++l) {
    const int e = E[l], h = H[l];
    const int waves   = (4 * h) / 16;            // one 16-col N-tile per wave
    const int gblocks = (waves * 32) / 256;      // 8 waves per block
    const int pblocks = (BATCH * h + 255) / 256;
    const size_t ldsbytes = (size_t)(e + h) * BATCH * sizeof(bf16);
    for (int s = 0; s < S_LEN; ++s) {
      const bf16* xt = layer_in[l] + (size_t)s * BATCH * e;
      const bf16* hp = (s == 0) ? nullptr
                                : layer_out[l] + (size_t)(s - 1) * BATCH * h;
      lstm_gates_wmma<<<gblocks, 256, ldsbytes, stream>>>(xt, e, hp, h,
                                                          wihb[l], whhb[l],
                                                          bihp[l], bhhp[l], gates);
      float* lastp = (l == 2) ? (last + (size_t)s * 256) : nullptr;
      lstm_pointwise<<<pblocks, 256, 0, stream>>>(gates, cstate,
                                                  layer_out[l] + (size_t)s * BATCH * h,
                                                  lastp, h, (s == 0) ? 1 : 0);
    }
  }

  // ---- head: (1024,256) @ (256,7)^T + b ----
  final_linear<<<(S_LEN * 7 + 255) / 256, 256, 0, stream>>>(last, Wlin, blin, out);
}
